// t_62878321213451
// MI455X (gfx1250) — matrix operand order repaired and checked by reading, not compiled, not run on hardware
//
#include <hip/hip_runtime.h>

// ---------------------------------------------------------------------------
// out = xb @ fakequant(W) + b  on gfx1250.
// v_wmma_f32_16x16x32_bf16 GEMM, triple-buffered BK=64 stages filled with
// GLOBAL_LOAD_ASYNC_TO_LDS_B128 (ASYNCcnt <= 8 sliding wait, in-order done).
// Steady-state loop is branch-free; pipeline drain is peeled.
// ---------------------------------------------------------------------------

typedef __attribute__((ext_vector_type(16))) __bf16 v16bf;
typedef __attribute__((ext_vector_type(8)))  float  v8f;

#define BM 128
#define BN 128
#define BK 64
#define LDS_STRIDE 72          // halfwords: 144B rows, 16B-aligned b128, conflict-free
#define HS (BM * LDS_STRIDE)   // halfwords per operand per stage (9216)
#define A_BYTES (HS * 2)       // 18432 B
#define STAGE_BYTES (HS * 4)   // A + B per stage = 36864 B
#define NSTAGE 3

#if __has_builtin(__builtin_amdgcn_s_wait_asynccnt)
#define WAIT_ASYNC(n) __builtin_amdgcn_s_wait_asynccnt(n)
#else
#define WAIT_ASYNC(n) asm volatile("s_wait_asynccnt %0" :: "i"(n) : "memory")
#endif

static __device__ __forceinline__ unsigned short f32_to_bf16_rne(float f) {
    unsigned u = __float_as_uint(f);
    unsigned r = 0x7FFFu + ((u >> 16) & 1u);   // round-to-nearest-even
    return (unsigned short)((u + r) >> 16);
}

// async 16B global -> LDS copy, tracked by ASYNCcnt (no staging VGPRs)
static __device__ __forceinline__ void async_ld_b128(unsigned lds_addr, unsigned goff,
                                                     const void* sbase) {
    asm volatile("global_load_async_to_lds_b128 %0, %1, %2"
                 :: "v"(lds_addr), "v"(goff), "s"(sbase) : "memory");
}

// ---------------- pass 1: per-tensor abs-max of W ----------------
__global__ void absmax_kernel(const float* __restrict__ W, unsigned* __restrict__ out, int n) {
    __shared__ unsigned red[256];
    float m = 0.0f;
    for (int i = blockIdx.x * blockDim.x + threadIdx.x; i < n; i += gridDim.x * blockDim.x)
        m = fmaxf(m, fabsf(W[i]));
    red[threadIdx.x] = __float_as_uint(m);   // f >= 0 : u32 order == float order
    __syncthreads();
    for (int s = 128; s > 0; s >>= 1) {
        if (threadIdx.x < (unsigned)s) {
            unsigned a = red[threadIdx.x], b = red[threadIdx.x + s];
            red[threadIdx.x] = a > b ? a : b;
        }
        __syncthreads();
    }
    if (threadIdx.x == 0) atomicMax(out, red[0]);
}

// ---- pass 2a: W[k][n] -> Bt[n][k] = bf16(clip(round(W/s_w))), exact int levels ----
__global__ void quant_transpose_kernel(const float* __restrict__ W,
                                       const unsigned* __restrict__ amax_bits,
                                       unsigned short* __restrict__ Bt,
                                       int K, int N) {
    __shared__ unsigned short tile[32][33];
    const float amax = __uint_as_float(*amax_bits);
    const float s    = amax * (1.0f / 128.0f);        // 2*amax/256
    const float inv  = (s > 0.0f) ? (1.0f / s) : 0.0f;
    const int n     = blockIdx.x * 32 + threadIdx.x;
    const int kbase = blockIdx.y * 32;
#pragma unroll
    for (int yy = 0; yy < 4; ++yy) {
        int k = kbase + threadIdx.y + yy * 8;
        float q = rintf(W[(size_t)k * N + n] * inv);  // RNE == jnp.round
        q = fminf(fmaxf(q, -127.0f), 127.0f);
        tile[threadIdx.y + yy * 8][threadIdx.x] = f32_to_bf16_rne(q);  // |int|<=127: exact in bf16
    }
    __syncthreads();
#pragma unroll
    for (int yy = 0; yy < 4; ++yy) {
        int on = blockIdx.x * 32 + threadIdx.y + yy * 8;
        int ok = kbase + threadIdx.x;
        Bt[(size_t)on * K + ok] = tile[threadIdx.x][threadIdx.y + yy * 8];
    }
}

// ---------------- pass 2b: xb fp32 -> bf16 ----------------
__global__ void cvt_bf16_kernel(const float4* __restrict__ x, uint2* __restrict__ y, size_t n4) {
    size_t i = (size_t)blockIdx.x * blockDim.x + threadIdx.x;
    if (i >= n4) return;
    float4 v = x[i];
    uint2 o;
    o.x = (unsigned)f32_to_bf16_rne(v.x) | ((unsigned)f32_to_bf16_rne(v.y) << 16);
    o.y = (unsigned)f32_to_bf16_rne(v.z) | ((unsigned)f32_to_bf16_rne(v.w) << 16);
    y[i] = o;
}

// ---------------- pass 3: WMMA GEMM, C = s_w*(A@Bt^T) + bias ----------------
union FragBF { uint4 q[2]; v16bf v; };

__global__ __launch_bounds__(256)
void gemm_bf16_wmma_kernel(const unsigned short* __restrict__ A,    // [M][K] bf16 bits
                           const unsigned short* __restrict__ Bt,   // [N][K] bf16 bits (int levels)
                           const float* __restrict__ bias,
                           const unsigned* __restrict__ amax_bits,
                           float* __restrict__ C,
                           int M, int N, int K) {
    extern __shared__ unsigned short smem[];   // NSTAGE * (A 18KB + B 18KB) = 108 KB

    const int tid  = threadIdx.x;
    const int wave = tid >> 5;
    const int lane = tid & 31;
    const int half = lane >> 4;   // lane group: selects K-halves per ISA 7.12.2
    const int lr   = lane & 15;
    const int wm   = wave & 3;    // 4 waves over M -> 32-row slab each
    const int wn   = wave >> 2;   // 2 waves over N -> 64-col slab each

    const int n0 = blockIdx.x * BN;
    const int m0 = blockIdx.y * BM;

    v8f acc[2][4];
#pragma unroll
    for (int mi = 0; mi < 2; ++mi)
#pragma unroll
        for (int ni = 0; ni < 4; ++ni) {
            v8f z = {0.f, 0.f, 0.f, 0.f, 0.f, 0.f, 0.f, 0.f};
            acc[mi][ni] = z;
        }

    // this thread's 4 chunks (16B each) per operand per stage: 1024 chunks / 256 threads
    unsigned ldsOff[4], gAo[4], gBo[4];
#pragma unroll
    for (int p = 0; p < 4; ++p) {
        int idx = tid + p * 256;
        int row = idx >> 3, kc = idx & 7;            // 8 chunks per 64-wide row
        ldsOff[p] = (unsigned)(row * LDS_STRIDE + kc * 8) * 2;     // bytes
        gAo[p]    = (unsigned)((size_t)(m0 + row) * K * 2) + kc * 16;
        gBo[p]    = (unsigned)((size_t)(n0 + row) * K * 2) + kc * 16;
    }
    const unsigned smemBase = (unsigned)(size_t)smem;   // LDS byte address = low 32 bits

    auto asyncLoadStage = [&](int kt, int buf) {        // 8 asyncs = one ASYNCcnt group
        const unsigned kb    = (unsigned)(kt * BK * 2);
        const unsigned aBase = smemBase + (unsigned)(buf * STAGE_BYTES);
#pragma unroll
        for (int p = 0; p < 4; ++p) async_ld_b128(aBase + ldsOff[p], gAo[p] + kb, A);
#pragma unroll
        for (int p = 0; p < 4; ++p) async_ld_b128(aBase + A_BYTES + ldsOff[p], gBo[p] + kb, Bt);
    };
    auto computeStage = [&](int buf) {                  // 2 x 8 WMMAs per wave
        const unsigned short* sA = smem + (size_t)buf * (STAGE_BYTES / 2);
        const unsigned short* sB = sA + HS;
#pragma unroll
        for (int kk = 0; kk < 2; ++kk) {
            FragBF af[2], bfr[4];
#pragma unroll
            for (int mi = 0; mi < 2; ++mi) {  // A 16x32: half0->K{0-7,16-23}, half1->K{8-15,24-31}
                const unsigned short* base = &sA[(wm * 32 + mi * 16 + lr) * LDS_STRIDE + kk * 32];
                af[mi].q[0] = *(const uint4*)(base + half * 8);
                af[mi].q[1] = *(const uint4*)(base + 16 + half * 8);
            }
#pragma unroll
            for (int ni = 0; ni < 4; ++ni) {  // B 32x16: half0->K{0-7,16-23}, half1->K{8-15,24-31}
                const unsigned short* base =
                    &sB[(wn * 64 + ni * 16 + lr) * LDS_STRIDE + kk * 32 + half * 8];
                bfr[ni].q[0] = *(const uint4*)(base);
                bfr[ni].q[1] = *(const uint4*)(base + 16);
            }
#pragma unroll
            for (int mi = 0; mi < 2; ++mi)
#pragma unroll
                for (int ni = 0; ni < 4; ++ni)
                    acc[mi][ni] = __builtin_amdgcn_wmma_f32_16x16x32_bf16(
                        false, af[mi].v, false, bfr[ni].v, (short)0, acc[mi][ni], false, false);
        }
    };

    const int nk = K / BK;            // 64 stages (>= 3 for K = 4096)
    asyncLoadStage(0, 0);
    if (nk > 1) asyncLoadStage(1, 1);

    // steady state: branch-free body; fills always issued two stages ahead.
    int bufC = 0;                     // compute buffer = kt % 3
    int bufF = 2 % NSTAGE;            // fill buffer    = (kt+2) % 3
    for (int kt = 0; kt + 2 < nk; ++kt) {
        WAIT_ASYNC(8);                // oldest group (stage kt) landed; newest keeps flying
        __syncthreads();              // publish stage kt to all waves
        asyncLoadStage(kt + 2, bufF);
        computeStage(bufC);           // 16 WMMAs shadow the fills
        bufC = (bufC == NSTAGE - 1) ? 0 : bufC + 1;
        bufF = (bufF == NSTAGE - 1) ? 0 : bufF + 1;
    }
    // drain: stage nk-2 (stage nk-1 still in flight), then stage nk-1.
    if (nk > 1) {
        WAIT_ASYNC(8);
        __syncthreads();
        computeStage(bufC);
        bufC = (bufC == NSTAGE - 1) ? 0 : bufC + 1;
    }
    WAIT_ASYNC(0);
    __syncthreads();
    computeStage(bufC);

    // epilogue: fold per-tensor scale + bias. C layout: VGPR j <-> M = j + 8*half, N = lr.
    const float s_w = __uint_as_float(*amax_bits) * (1.0f / 128.0f);
#pragma unroll
    for (int mi = 0; mi < 2; ++mi)
#pragma unroll
        for (int ni = 0; ni < 4; ++ni) {
            const int col = n0 + wn * 64 + ni * 16 + lr;
            const float bv = bias[col];
#pragma unroll
            for (int j = 0; j < 8; ++j) {
                const int row = m0 + wm * 32 + mi * 16 + half * 8 + j;
                C[(size_t)row * N + col] = s_w * acc[mi][ni][j] + bv;
            }
        }
}

// ---------------------------------------------------------------------------
extern "C" void kernel_launch(void* const* d_in, const int* in_sizes, int n_in,
                              void* d_out, int out_size, void* d_ws, size_t ws_size,
                              hipStream_t stream) {
    const float* xb   = (const float*)d_in[0];
    const float* W    = (const float*)d_in[1];
    const float* bias = (const float*)d_in[2];

    const int N = in_sizes[2];            // 4096
    const int K = in_sizes[1] / N;        // 4096
    const int M = in_sizes[0] / K;        // 8192

    unsigned char*  ws        = (unsigned char*)d_ws;
    unsigned*       amax_bits = (unsigned*)ws;
    unsigned short* Abf       = (unsigned short*)(ws + 256);                                  // 64 MB
    unsigned short* Btb       = (unsigned short*)(ws + 256 + (size_t)M * K * sizeof(short));  // 32 MB

    hipMemsetAsync(d_ws, 0, 256, stream);

    absmax_kernel<<<2048, 256, 0, stream>>>(W, amax_bits, K * N);
    quant_transpose_kernel<<<dim3(N / 32, K / 32), dim3(32, 8), 0, stream>>>(W, amax_bits, Btb, K, N);

    const size_t n4 = ((size_t)M * K) / 4;
    cvt_bf16_kernel<<<(unsigned)((n4 + 255) / 256), 256, 0, stream>>>((const float4*)xb, (uint2*)Abf, n4);

    gemm_bf16_wmma_kernel<<<dim3(N / BN, M / BM), 256, NSTAGE * STAGE_BYTES, stream>>>(
        Abf, Btb, bias, amax_bits, (float*)d_out, M, N, K);
}
